// RNN_86904368267679
// MI455X (gfx1250) — matrix operand order repaired and checked by reading, not compiled, not run on hardware
//
#include <hip/hip_runtime.h>

// ---------------------------------------------------------------------------
// GRU stack (L=4, H=1024, F=512, T=4096) for gfx1250.
//   Phase 0: convert weights/batch fp32 -> f16 (workspace)
//   Phase 1: Gi0 = batch @ W_ih0^T + b_ih0 as dense WMMA GEMM (no recurrence)
//   Phase 2: persistent recurrent kernel, 64 WGs x 384 thr, WMMA matvecs,
//            async global->LDS vector staging, 1 grid barrier per layer.
// ---------------------------------------------------------------------------

typedef __attribute__((ext_vector_type(16))) _Float16 v16h;
typedef __attribute__((ext_vector_type(8)))  float    v8f;
typedef __attribute__((ext_vector_type(4)))  int      v4i;

#define HD     1024
#define G3H    3072
#define FDIM   512
#define TSTEPS 4096
#define NWG    64

#if defined(__gfx1250__) && defined(__has_builtin)
# if __has_builtin(__builtin_amdgcn_global_load_async_to_lds_b128) && \
     __has_builtin(__builtin_amdgcn_s_wait_asynccnt)
#  define USE_ASYNC_LDS 1
# endif
#endif
#ifndef USE_ASYNC_LDS
# define USE_ASYNC_LDS 0
#endif

struct __align__(32) H16 { uint4 u0, u1; };

// A-fragment (16x32 f16, row-major source): lane L<16 holds row L, K {0..7,16..23};
// lane L>=16 holds row L-16, K {8..15,24..31}.  Two 16B loads.
__device__ inline v16h ld_afrag(const _Float16* __restrict__ row, int k0, int lane) {
    const int aoff = (lane < 16) ? 0 : 8;
    union { H16 h; v16h v; } u;
    u.h.u0 = *(const uint4*)(row + k0 + aoff);
    u.h.u1 = *(const uint4*)(row + k0 + aoff + 16);
    return u.v;
}

// B-fragment: lanes 0-15 -> K {0..7,16..23}, lanes 16-31 -> K {8..15,24..31}
__device__ inline v16h ld_b16(const _Float16* __restrict__ p) {
    union { H16 h; v16h v; } u;
    u.h.u0 = *(const uint4*)(p);
    u.h.u1 = *(const uint4*)(p + 16);
    return u.v;
}

// 16B global -> LDS transfer (async on gfx1250, vector copy otherwise).
__device__ inline void cp16_g2l(_Float16* lds_dst, const _Float16* gsrc) {
#if USE_ASYNC_LDS
    v4i* gg = (v4i*)const_cast<_Float16*>(gsrc);
    v4i* ll = (v4i*)lds_dst;
    __builtin_amdgcn_global_load_async_to_lds_b128(
        (__attribute__((address_space(1))) v4i*)gg,
        (__attribute__((address_space(3))) v4i*)ll, 0, 0);
#else
    *(uint4*)lds_dst = *(const uint4*)gsrc;
#endif
}
__device__ inline void cp16_wait() {
#if USE_ASYNC_LDS
    __builtin_amdgcn_s_wait_asynccnt(0);
#endif
}

// ---------------------------------------------------------------------------
__global__ void cvt_f32_f16(const float* __restrict__ in, _Float16* __restrict__ out, int n) {
    for (int i = blockIdx.x * blockDim.x + threadIdx.x; i < n; i += gridDim.x * blockDim.x)
        out[i] = (_Float16)in[i];
}

// ---------------------------------------------------------------------------
// Gi0[t, 3H] = batch[t,:] @ W_ih0^T + b_ih0.  One 16x16 tile per wave, K=512.
__global__ __launch_bounds__(256) void gemm_gi0(
    const _Float16* __restrict__ Xh,     // (4096, 512)
    const _Float16* __restrict__ Wih0h,  // (3072, 512) row-major
    const float*    __restrict__ b_ih0,  // (3072)
    float*          __restrict__ Gi0)    // (4096, 3072)
{
    const int lane   = threadIdx.x & 31;
    const int wv     = threadIdx.x >> 5;
    const int tileid = blockIdx.x * 8 + wv;          // 256*192 = 49152 tiles
    const int ti = tileid / 192, tj = tileid % 192;
    const int t0 = ti * 16, n0 = tj * 16;

    const _Float16* arow = Xh    + (size_t)(t0 + (lane & 15)) * FDIM;
    const _Float16* brow = Wih0h + (size_t)(n0 + (lane & 15)) * FDIM;
    const int boff = (lane < 16) ? 0 : 8;

    v8f c = {};
#pragma unroll 4
    for (int k0 = 0; k0 < FDIM; k0 += 32) {
        v16h A = ld_afrag(arow, k0, lane);
        v16h B = ld_b16(brow + k0 + boff);
        c = __builtin_amdgcn_wmma_f32_16x16x32_f16(false, A, false, B, (short)0, c, false, false);
    }
    // D layout: VGPR r -> (M=r, N=lane) for lanes 0-15, (M=8+r, N=lane-16) for lanes 16-31
    const int   nc   = n0 + (lane & 15);
    const float bias = b_ih0[nc];
    const int   mb   = t0 + ((lane < 16) ? 0 : 8);
#pragma unroll
    for (int r = 0; r < 8; ++r)
        Gi0[(size_t)(mb + r) * G3H + nc] = c[r] + bias;
}

// ---------------------------------------------------------------------------
// Two-counter monotonic grid barrier (counters zeroed by memset each launch).
__device__ inline void grid_barrier(unsigned* cnt, int iter) {
    __syncthreads();
    if (threadIdx.x == 0) {
        __threadfence();
        unsigned* c = cnt + (iter & 1);
        atomicAdd(c, 1u);
        const unsigned target = (unsigned)NWG * (unsigned)((iter >> 1) + 1);
        while (__hip_atomic_load(c, __ATOMIC_ACQUIRE, __HIP_MEMORY_SCOPE_AGENT) < target)
            __builtin_amdgcn_s_sleep(2);
    }
    __syncthreads();
}

// ---------------------------------------------------------------------------
// Persistent recurrent kernel.  WG w owns h-positions [16w, 16w+16).
// 12 waves = 6 tiles (gate r/z/n  x  matrix ih/hh) x 2 K-halves (512 each).
__global__ __launch_bounds__(384) void rnn_persist(
    const _Float16* __restrict__ Whh0h,  // (3072,1024)
    const _Float16* __restrict__ Wihh,   // (3,3072,1024)
    const _Float16* __restrict__ Whhh,   // (3,3072,1024)
    const float*    __restrict__ Gi0,    // (4096,3072) incl. b_ih0
    const float*    __restrict__ b_hh0,
    const float*    __restrict__ b_ih,   // (3,3072)
    const float*    __restrict__ b_hh,   // (3,3072)
    const float*    __restrict__ fc1_w, const float* __restrict__ fc1_b,
    const float*    __restrict__ fc2_w, const float* __restrict__ fc2_b,
    float*          __restrict__ hstate, // (2,4,1024) fp32, zeroed at launch
    _Float16*       __restrict__ hhalf,  // (2,4,1024) f16,  zeroed at launch
    unsigned*       __restrict__ cnt,
    float*          __restrict__ out)    // (4096)
{
    const int w     = blockIdx.x;
    const int tid   = threadIdx.x;
    const int lane  = tid & 31;
    const int wv    = tid >> 5;      // 0..11
    const int tile  = wv >> 1;       // 0..5
    const int khalf = wv & 1;
    const int matr  = tile / 3;      // 0 = ih, 1 = hh
    const int gate  = tile % 3;      // r, z, n
    const int p0    = w * 16;
    const int rowb  = gate * HD + p0;
    const int kbase = khalf * 512;

    __shared__ __align__(32) _Float16 shv[2][HD];  // [0]=inp, [1]=h[l] (f16)
    __shared__ float part[12][16];
    __shared__ float shfc[32];

    for (int t = 0; t < TSTEPS; ++t) {
        const int pr = t & 1, pw = pr ^ 1;
        for (int l = 0; l < 4; ++l) {
            // Stage broadcast vectors into LDS: 128 x 16B chunks per vector.
            // Threads 0-127 -> shv[1] (h[l], t-1); threads 128-255 -> shv[0] (inp).
            if (tid < 128) {
                const _Float16* hsrc = hhalf + (size_t)(pr * 4 + l) * HD;
                cp16_g2l(&shv[1][tid * 8], hsrc + tid * 8);
            } else if (l > 0 && tid < 256) {
                const int ci = tid - 128;
                const _Float16* isrc = hhalf + (size_t)(pw * 4 + (l - 1)) * HD;
                cp16_g2l(&shv[0][ci * 8], isrc + ci * 8);
            }
            cp16_wait();
            __syncthreads();

            v8f c = {};
            if (!(matr == 0 && l == 0)) {   // layer-0 ih path precomputed in Gi0
                const _Float16* W = (matr == 0)
                    ? (Wihh + (size_t)(l - 1) * G3H * HD)
                    : ((l == 0) ? Whh0h : (Whhh + (size_t)(l - 1) * G3H * HD));
                const _Float16* wrow = W + (size_t)(rowb + (lane & 15)) * HD;
                const _Float16* vec  = &shv[matr ? 1 : 0][0];
                const int boff = (lane < 16) ? 0 : 8;
#pragma unroll 4
                for (int kk = 0; kk < 512; kk += 32) {
                    const int k0 = kbase + kk;
                    v16h A = ld_afrag(wrow, k0, lane);
                    v16h B = ld_b16(vec + k0 + boff);   // all 16 columns == h chunk
                    c = __builtin_amdgcn_wmma_f32_16x16x32_f16(false, A, false, B,
                                                               (short)0, c, false, false);
                }
            }
            // column N=0 carries the matvec result: lane 0 -> M 0..7, lane 16 -> M 8..15
            if (lane == 0 || lane == 16) {
                const int mo = (lane == 16) ? 8 : 0;
#pragma unroll
                for (int r = 0; r < 8; ++r) part[wv][mo + r] = c[r];
            }
            __syncthreads();

            // GRU gate math for this WG's 16 positions.
            if (tid < 16) {
                const int j = tid, p = p0 + j;
                float gir, giz, gin;
                if (l == 0) {
                    const float* g = Gi0 + (size_t)t * G3H;
                    gir = g[p]; giz = g[HD + p]; gin = g[2 * HD + p];
                } else {
                    const float* bi = b_ih + (size_t)(l - 1) * G3H;
                    gir = part[0][j] + part[1][j] + bi[p];
                    giz = part[2][j] + part[3][j] + bi[HD + p];
                    gin = part[4][j] + part[5][j] + bi[2 * HD + p];
                }
                const float* bh = (l == 0) ? b_hh0 : (b_hh + (size_t)(l - 1) * G3H);
                const float ghr = part[6][j]  + part[7][j]  + bh[p];
                const float ghz = part[8][j]  + part[9][j]  + bh[HD + p];
                const float ghn = part[10][j] + part[11][j] + bh[2 * HD + p];
                const float r = 1.f / (1.f + __expf(-(gir + ghr)));
                const float z = 1.f / (1.f + __expf(-(giz + ghz)));
                const float n = tanhf(gin + r * ghn);
                const float hp = hstate[(pr * 4 + l) * HD + p];
                const float hn = (1.f - z) * n + z * hp;
                hstate[(pw * 4 + l) * HD + p] = hn;
                hhalf [(pw * 4 + l) * HD + p] = (_Float16)hn;
                __threadfence();
            }
            grid_barrier(cnt, t * 4 + l);
        }

        // MLP head on WG0 (overlaps other WGs starting layer 0 of t+1).
        if (w == 0) {
            const float* htop = hstate + (pw * 4 + 3) * HD;
            for (int r = wv; r < 32; r += 12) {
                float s = 0.f;
                for (int k = lane; k < HD; k += 32) s += fc1_w[r * HD + k] * htop[k];
                for (int off = 16; off > 0; off >>= 1) s += __shfl_down(s, off, 32);
                if (lane == 0) shfc[r] = fmaxf(s + fc1_b[r], 0.f);
            }
            __syncthreads();
            if (wv == 0) {
                float s = shfc[lane] * fc2_w[lane];
                for (int off = 16; off > 0; off >>= 1) s += __shfl_down(s, off, 32);
                if (lane == 0) out[t] = s + fc2_b[0];
            }
            __syncthreads();
        }
    }
}

// ---------------------------------------------------------------------------
extern "C" void kernel_launch(void* const* d_in, const int* in_sizes, int n_in,
                              void* d_out, int out_size, void* d_ws, size_t ws_size,
                              hipStream_t stream) {
    (void)in_sizes; (void)n_in; (void)out_size; (void)ws_size;
    const float* batch = (const float*)d_in[0];
    const float* W_ih0 = (const float*)d_in[1];
    const float* W_hh0 = (const float*)d_in[2];
    const float* b_ih0 = (const float*)d_in[3];
    const float* b_hh0 = (const float*)d_in[4];
    const float* W_ih  = (const float*)d_in[5];
    const float* W_hh  = (const float*)d_in[6];
    const float* b_ih  = (const float*)d_in[7];
    const float* b_hh  = (const float*)d_in[8];
    const float* fc1_w = (const float*)d_in[9];
    const float* fc1_b = (const float*)d_in[10];
    const float* fc2_w = (const float*)d_in[11];
    const float* fc2_b = (const float*)d_in[12];
    float* out = (float*)d_out;

    char* ws = (char*)d_ws;
    size_t o = 0;
    unsigned*  cnt    = (unsigned*)(ws + o);  o += 256;
    float*     hstate = (float*)(ws + o);     o += 2u * 4u * HD * 4u;       // 32 KB
    _Float16*  hhalf  = (_Float16*)(ws + o);  o += 2u * 4u * HD * 2u;       // 16 KB
    const size_t zero_bytes = o;
    _Float16*  Xh     = (_Float16*)(ws + o);  o += (size_t)TSTEPS * FDIM * 2u;
    _Float16*  Wih0h  = (_Float16*)(ws + o);  o += (size_t)G3H * FDIM * 2u;
    _Float16*  Whh0h  = (_Float16*)(ws + o);  o += (size_t)G3H * HD * 2u;
    _Float16*  Wihh   = (_Float16*)(ws + o);  o += 3u * (size_t)G3H * HD * 2u;
    _Float16*  Whhh   = (_Float16*)(ws + o);  o += 3u * (size_t)G3H * HD * 2u;
    float*     Gi0    = (float*)(ws + o);     o += (size_t)TSTEPS * G3H * 4u;

    // Zero barrier counters + hidden state (deterministic across graph replays).
    (void)hipMemsetAsync(d_ws, 0, zero_bytes, stream);

    cvt_f32_f16<<<1024, 256, 0, stream>>>(batch, Xh,    TSTEPS * FDIM);
    cvt_f32_f16<<<1024, 256, 0, stream>>>(W_ih0, Wih0h, G3H * FDIM);
    cvt_f32_f16<<<1024, 256, 0, stream>>>(W_hh0, Whh0h, G3H * HD);
    cvt_f32_f16<<<2048, 256, 0, stream>>>(W_ih,  Wihh,  3 * G3H * HD);
    cvt_f32_f16<<<2048, 256, 0, stream>>>(W_hh,  Whhh,  3 * G3H * HD);

    gemm_gi0<<<6144, 256, 0, stream>>>(Xh, Wih0h, b_ih0, Gi0);

    rnn_persist<<<NWG, 384, 0, stream>>>(Whh0h, Wihh, Whhh, Gi0,
                                         b_hh0, b_ih, b_hh,
                                         fc1_w, fc1_b, fc2_w, fc2_b,
                                         hstate, hhalf, cnt, out);
}
